// NAM_695784702164
// MI455X (gfx1250) — matrix operand order repaired and checked by reading, not compiled, not run on hardware
//
#include <hip/hip_runtime.h>
#include <hip/hip_bf16.h>

typedef _Float16 v16h __attribute__((ext_vector_type(16)));
typedef _Float16 v8h  __attribute__((ext_vector_type(8)));
typedef float    v8f  __attribute__((ext_vector_type(8)));

#define USE_ASYNC_COPY 1   // global_load_async_to_lds_b128 via inline asm

// Packed fp16 weight layout (per feature, 81920 halves):
//   L2 @ 0      (8192 halves),  L3 @ 8192  (32768),
//   L4 @ 40960  (32768),        L5 @ 73728 (8192)
// Each 32x16 (KxN) tile is 512 halves, lane-major: lane L holds 16 contiguous
// halves = column (L%16), K-range (L<16 ? 0..15 : 16..31) -> B fragment is two
// contiguous ds_load_b128 per lane.
#define FEAT_STRIDE 81920

// ---------------------------------------------------------------------------
// Prep: convert W2..W5 fp32 -> packed/swizzled fp16 in global scratch.
// One chunk = 8 halves = one lane-half of a tile; 64 chunks per tile.
// ---------------------------------------------------------------------------
__global__ __launch_bounds__(256) void prep_weights_kernel(
    const float* __restrict__ W2, const float* __restrict__ W3,
    const float* __restrict__ W4, const float* __restrict__ W5,
    _Float16* __restrict__ packed) {
  const int gchunk = blockIdx.x * 256 + threadIdx.x;  // global chunk id
  const int tile = gchunk >> 6;                       // 64 chunks / tile
  const int cc   = gchunk & 63;
  const int f    = tile / 160;                        // 160 tiles / feature
  const int tf   = tile - f * 160;

  const float* src; int N, NT, tIn; size_t dstBase;
  if (tf < 16)       { src = W2 + (size_t)f * 64  * 128; N = 128; NT = 8;  tIn = tf;       dstBase = 0;     }
  else if (tf < 80)  { src = W3 + (size_t)f * 128 * 256; N = 256; NT = 16; tIn = tf - 16;  dstBase = 8192;  }
  else if (tf < 144) { src = W4 + (size_t)f * 256 * 128; N = 128; NT = 8;  tIn = tf - 80;  dstBase = 40960; }
  else               { src = W5 + (size_t)f * 128 * 64;  N = 64;  NT = 4;  tIn = tf - 144; dstBase = 73728; }

  const int kt = tIn / NT, nt = tIn % NT;
  const int lane = cc >> 1, part = cc & 1;
  const int nc = lane & 15;
  const int kr = ((lane >> 4) << 4) + part * 8;
  const float* sp = src + (size_t)(kt * 32 + kr) * N + nt * 16 + nc;
  v8h v;
#pragma unroll
  for (int j = 0; j < 8; ++j) v[j] = (_Float16)sp[(size_t)j * N];
  *(v8h*)(packed + (size_t)f * FEAT_STRIDE + dstBase + (size_t)tIn * 512 + cc * 8) = v;
}

// ---------------------------------------------------------------------------
// Staging: packed fp16 global -> LDS (16B chunks).
// ---------------------------------------------------------------------------
__device__ inline void stage_packed(const _Float16* __restrict__ g,
                                    _Float16* __restrict__ l,
                                    int halfs, int tid) {
#if USE_ASYNC_COPY
  const unsigned long long base = (unsigned long long)(uintptr_t)g;
  for (int i = tid; i < (halfs >> 3); i += 256) {
    const unsigned loff = (unsigned)(uintptr_t)(l + i * 8);  // LDS byte addr
    const unsigned goff = (unsigned)(i * 16);                // byte offset
    asm volatile("global_load_async_to_lds_b128 %0, %1, %2"
                 :: "v"(loff), "v"(goff), "s"(base)
                 : "memory");
  }
#else
  for (int i = tid; i < (halfs >> 3); i += 256)
    *(v8h*)(l + i * 8) = *(const v8h*)(g + i * 8);
#endif
}

__device__ inline void wait_stage() {
#if USE_ASYNC_COPY
  asm volatile("s_wait_asynccnt 0" ::: "memory");
#endif
}

// ---------------------------------------------------------------------------
// Fallback staging: fp32 global -> swizzled fp16 LDS, column-gather so each
// thread writes one contiguous ds_store_b128.
// ---------------------------------------------------------------------------
template <int K, int N>
__device__ inline void stage_convert(const float* __restrict__ src,
                                     _Float16* __restrict__ dst, int tid) {
  constexpr int NT = N / 16;
  constexpr int CH = K * N / 8;
  for (int c = tid; c < CH; c += 256) {
    const int t = c >> 6, cc = c & 63;
    const int kt = t / NT, nt = t % NT;  // NT power of two -> shifts
    const int lane = cc >> 1, part = cc & 1;
    const int nc = lane & 15;
    const int kr = ((lane >> 4) << 4) + part * 8;
    const float* sp = src + (kt * 32 + kr) * N + nt * 16 + nc;
    v8h v;
#pragma unroll
    for (int j = 0; j < 8; ++j) v[j] = (_Float16)sp[j * N];
    *(v8h*)(dst + t * 512 + cc * 8) = v;
  }
}

// ---------------------------------------------------------------------------
// One dense layer per wave: actOut[16 x N] = relu(actIn[16 x K] * W + bias).
// ---------------------------------------------------------------------------
template <int K, int N>
__device__ inline void mlp_layer(const _Float16* __restrict__ actIn,
                                 _Float16* __restrict__ actOut,
                                 const _Float16* __restrict__ wLds,
                                 const float* __restrict__ biasG,
                                 int lane, int rowBase) {
  constexpr int KT = K / 32, NT = N / 16;
  const int m  = lane & 15;
  const int hi = lane >> 4;

  v16h a[KT];
  const _Float16* arow = actIn + (rowBase + m) * K;
#pragma unroll
  for (int kt = 0; kt < KT; ++kt) {
    const _Float16* p = arow + kt * 32 + hi * 16;
    v8h lo = *(const v8h*)p;
    v8h hh = *(const v8h*)(p + 8);
    a[kt] = __builtin_shufflevector(lo, hh, 0, 1, 2, 3, 4, 5, 6, 7,
                                            8, 9, 10, 11, 12, 13, 14, 15);
  }

#pragma unroll
  for (int nt = 0; nt < NT; ++nt) {
    v8f c = {};
#pragma unroll
    for (int kt = 0; kt < KT; ++kt) {
      const _Float16* wp = wLds + (kt * NT + nt) * 512 + lane * 16;
      v8h blo = *(const v8h*)wp;
      v8h bhi = *(const v8h*)(wp + 8);
      v16h b = __builtin_shufflevector(blo, bhi, 0, 1, 2, 3, 4, 5, 6, 7,
                                                 8, 9, 10, 11, 12, 13, 14, 15);
      c = __builtin_amdgcn_wmma_f32_16x16x32_f16(
          false, a[kt], false, b, (short)0, c, false, false);
    }
    const int col = nt * 16 + m;
    const float bias = biasG[col];
#pragma unroll
    for (int r = 0; r < 8; ++r) {
      float v = c[r] + bias;
      v = v > 0.0f ? v : 0.0f;
      actOut[(rowBase + hi * 8 + r) * N + col] = (_Float16)v;
    }
  }
}

// ---------------------------------------------------------------------------
// Main kernel: block = (128-row batch tile) x (feature f); 8 wave32s.
// Double-buffered weights: stage layer L+1 while computing layer L.
// LDS: wbuf0|wbuf1|actA|actB = 4 x 64 KB = 256 KB.
// ---------------------------------------------------------------------------
template <bool PACKED>
__global__ __launch_bounds__(256) void mlp_wmma_kernel(
    const float* __restrict__ x,
    const float* __restrict__ W1, const float* __restrict__ b1,
    const float* __restrict__ W2, const float* __restrict__ b2,
    const float* __restrict__ W3, const float* __restrict__ b3,
    const float* __restrict__ W4, const float* __restrict__ b4,
    const float* __restrict__ W5, const float* __restrict__ b5,
    const float* __restrict__ W6, const float* __restrict__ b6,
    const _Float16* __restrict__ packed, float* __restrict__ acc, int F) {
  extern __shared__ char smem[];
  _Float16* wbuf0 = (_Float16*)smem;
  _Float16* wbuf1 = (_Float16*)(smem + 65536);
  _Float16* actA  = (_Float16*)(smem + 131072);
  _Float16* actB  = (_Float16*)(smem + 196608);

  const int f     = blockIdx.y;
  const int rowG0 = blockIdx.x * 128;
  const int tid   = threadIdx.x;
  const int lane  = tid & 31;
  const int rowBase = (tid >> 5) * 16;
  const _Float16* pw = packed + (size_t)f * FEAT_STRIDE;

  // Kick W2 staging, then do layer 1 while it is in flight.
  if (PACKED) stage_packed(pw + 0, wbuf0, 8192, tid);
  else        stage_convert<64, 128>(W2 + (size_t)f * 64 * 128, wbuf0, tid);

  {  // ---- Layer 1: scalar -> 64, elementwise ----
    const float* w1f = W1 + f * 64;
    const float* b1f = b1 + f * 64;
    for (int e = tid; e < 128 * 64; e += 256) {
      const int row = e >> 6, i = e & 63;
      const float xv = x[(rowG0 + row) * F + f];
      float h = xv * w1f[i] + b1f[i];
      actA[row * 64 + i] = (_Float16)(h > 0.0f ? h : 0.0f);
    }
  }
  if (PACKED) wait_stage();
  __syncthreads();

  // ---- Layer 2 (wbuf0) | prefetch W3 -> wbuf1 ----
  if (PACKED) stage_packed(pw + 8192, wbuf1, 32768, tid);
  else        stage_convert<128, 256>(W3 + (size_t)f * 128 * 256, wbuf1, tid);
  mlp_layer<64, 128>(actA, actB, wbuf0, b2 + f * 128, lane, rowBase);
  if (PACKED) wait_stage();
  __syncthreads();

  // ---- Layer 3 (wbuf1) | prefetch W4 -> wbuf0 ----
  if (PACKED) stage_packed(pw + 40960, wbuf0, 32768, tid);
  else        stage_convert<256, 128>(W4 + (size_t)f * 256 * 128, wbuf0, tid);
  mlp_layer<128, 256>(actB, actA, wbuf1, b3 + f * 256, lane, rowBase);
  if (PACKED) wait_stage();
  __syncthreads();

  // ---- Layer 4 (wbuf0) | prefetch W5 -> wbuf1 ----
  if (PACKED) stage_packed(pw + 73728, wbuf1, 8192, tid);
  else        stage_convert<128, 64>(W5 + (size_t)f * 128 * 64, wbuf1, tid);
  mlp_layer<256, 128>(actA, actB, wbuf0, b4 + f * 128, lane, rowBase);
  if (PACKED) wait_stage();
  __syncthreads();

  // ---- Layer 5 (wbuf1) ----
  mlp_layer<128, 64>(actB, actA, wbuf1, b5 + f * 64, lane, rowBase);
  __syncthreads();

  // ---- Layer 6: 64 -> 1, accumulate over features ----
  if (tid < 128) {
    const float* w6f = W6 + f * 64;
    float s = b6[f];
#pragma unroll 8
    for (int i = 0; i < 64; ++i)
      s += (float)actA[tid * 64 + i] * w6f[i];
    atomicAdd(&acc[rowG0 + tid], s);
  }
}

__global__ void init_acc_kernel(float* __restrict__ acc, int n) {
  int i = blockIdx.x * blockDim.x + threadIdx.x;
  if (i < n) acc[i] = 0.0f;
}

__global__ void finalize_kernel(const float* __restrict__ acc,
                                const float* __restrict__ bias,
                                float* __restrict__ out, int n) {
  int i = blockIdx.x * blockDim.x + threadIdx.x;
  if (i < n) {
    const float s = acc[i] + bias[0];
    out[i] = 1.0f / (1.0f + expf(-s));
  }
}

extern "C" void kernel_launch(void* const* d_in, const int* in_sizes, int n_in,
                              void* d_out, int out_size, void* d_ws, size_t ws_size,
                              hipStream_t stream) {
  const float* x    = (const float*)d_in[0];
  const float* W1   = (const float*)d_in[1];
  const float* b1   = (const float*)d_in[2];
  const float* W2   = (const float*)d_in[3];
  const float* b2   = (const float*)d_in[4];
  const float* W3   = (const float*)d_in[5];
  const float* b3   = (const float*)d_in[6];
  const float* W4   = (const float*)d_in[7];
  const float* b4   = (const float*)d_in[8];
  const float* W5   = (const float*)d_in[9];
  const float* b5   = (const float*)d_in[10];
  const float* W6   = (const float*)d_in[11];
  const float* b6   = (const float*)d_in[12];
  const float* bias = (const float*)d_in[13];
  float* out = (float*)d_out;

  const int F = 128;
  const int B = in_sizes[0] / F;   // 2048
  const int btiles = B / 128;      // 16
  const int smemBytes = 4 * 64 * 1024;  // 256 KB (<= 320 KB/WGP)

  float* acc = (float*)d_ws;
  _Float16* pw = (_Float16*)((char*)d_ws + 8192);
  const size_t needWs = 8192 + (size_t)F * FEAT_STRIDE * sizeof(_Float16);
  const bool usePacked = ws_size >= needWs;

  init_acc_kernel<<<(B + 255) / 256, 256, 0, stream>>>(acc, B);

  if (usePacked) {
    (void)hipFuncSetAttribute((const void*)mlp_wmma_kernel<true>,
                              hipFuncAttributeMaxDynamicSharedMemorySize,
                              smemBytes);
    // 128 features * 160 tiles * 64 chunks / 256 threads = 5120 blocks
    prep_weights_kernel<<<5120, 256, 0, stream>>>(W2, W3, W4, W5, pw);
    mlp_wmma_kernel<true><<<dim3(btiles, F), 256, smemBytes, stream>>>(
        x, W1, b1, W2, b2, W3, b3, W4, b4, W5, b5, W6, b6, pw, acc, F);
  } else {
    (void)hipFuncSetAttribute((const void*)mlp_wmma_kernel<false>,
                              hipFuncAttributeMaxDynamicSharedMemorySize,
                              smemBytes);
    mlp_wmma_kernel<false><<<dim3(btiles, F), 256, smemBytes, stream>>>(
        x, W1, b1, W2, b2, W3, b3, W4, b4, W5, b5, W6, b6, pw, acc, F);
  }

  finalize_kernel<<<(B + 255) / 256, 256, 0, stream>>>(acc, bias, out, B);
}
